// Model_64982855188964
// MI455X (gfx1250) — compile-verified, matrix operand order checked by reading, not run on hardware
//
#include <hip/hip_runtime.h>

typedef __attribute__((ext_vector_type(8))) int v8i;

#define IMG_PER_BLK 64
#define BLK_THREADS 128
#define HSTRIDE     1936           // bytes per hb row (1920 used + pad, bank-friendly)
#define K_PAD       1920           // 30 * 64
#define K_REAL      1875
#define KSTEPS      30
#define XBYTES      784            // 28*28 per image

__device__ __forceinline__ signed char sgn8(float v) {
    return (signed char)((v > 0.0f) - (v < 0.0f));
}
__device__ __forceinline__ float sgnf(float v) {
    return (v > 0.0f) ? 1.0f : ((v < 0.0f) ? -1.0f : 0.0f);
}

// ---------------------------------------------------------------------------
// Kernel 0: binarize w_fc [10,1875] into WMMA B-fragment layout for
// V_WMMA_I32_16X16X64_IU8.  Per k-step kk (64 K values), per lane L (32),
// 8 dwords of 4 int8 each:  N = L%16 (cols >=10 zero), K-half = (L/16)*8,
// dword g covers bytes K = kk*64 + (g/2)*16 + khalf + (g%2)*4 .. +3.
// Total: 30*32*8 ints = 30 KB in workspace.
// ---------------------------------------------------------------------------
__global__ void bnn_pack_wfc(const float* __restrict__ wfc, int* __restrict__ wsB) {
    int idx = blockIdx.x * blockDim.x + threadIdx.x;
    const int total = KSTEPS * 32 * 8;
    for (int i = idx; i < total; i += gridDim.x * blockDim.x) {
        int kk   = i >> 8;          // / 256
        int rem  = i & 255;
        int lane = rem >> 3;
        int g    = rem & 7;
        int N     = lane & 15;
        int khalf = (lane >> 4) * 8;
        int kbase = kk * 64 + (g >> 1) * 16 + khalf + (g & 1) * 4;
        unsigned int packed = 0;
#pragma unroll
        for (int j = 0; j < 4; ++j) {
            int K = kbase + j;
            signed char s = 0;
            if (N < 10 && K < K_REAL) s = sgn8(wfc[N * K_REAL + K]);
            packed |= ((unsigned int)(unsigned char)s) << (8 * j);
        }
        wsB[i] = (int)packed;
    }
}

// ---------------------------------------------------------------------------
// Fused kernel: binarize -> conv4x4(+bias,sign) -> FC via WMMA IU8 -> sign.
// 128 threads (4 waves), 64 images per block.
// ---------------------------------------------------------------------------
__global__ __launch_bounds__(BLK_THREADS)
void bnn_fused(const float* __restrict__ x,
               const float* __restrict__ wconv,
               const float* __restrict__ bconv,
               const float* __restrict__ bfc,
               const int*   __restrict__ wsB,
               float*       __restrict__ out,
               int nImg) {
    extern __shared__ char lds[];
    signed char* hsh = (signed char*)lds;                          // 64 * 1936
    signed char* xsh = (signed char*)(lds + IMG_PER_BLK * HSTRIDE); // 64 * 784

    const int  tid      = threadIdx.x;
    const long blockImg = (long)blockIdx.x * IMG_PER_BLK;

    // Hint the (L2-resident) B-fragment table into cache early.
    __builtin_prefetch(wsB + (tid & 31) * 8, 0, 0);

    // -------- Phase A: binarize x into LDS (int8 +/-1), float4 streaming ----
    {
        const float4* xv   = (const float4*)(x + blockImg * XBYTES);
        int*          xshi = (int*)xsh;
        const int nvec = IMG_PER_BLK * XBYTES / 4;   // 12544
        for (int i = tid; i < nvec; i += BLK_THREADS) {
            float4 v = xv[i];
            unsigned int p =  (unsigned int)(unsigned char)sgn8(v.x)
                           | ((unsigned int)(unsigned char)sgn8(v.y) << 8)
                           | ((unsigned int)(unsigned char)sgn8(v.z) << 16)
                           | ((unsigned int)(unsigned char)sgn8(v.w) << 24);
            xshi[i] = (int)p;
        }
    }

    // Binarized conv weights + float biases in registers.
    signed char wcb[3][16];
    float       bc[3];
#pragma unroll
    for (int ch = 0; ch < 3; ++ch) {
        bc[ch] = bconv[ch];
#pragma unroll
        for (int k = 0; k < 16; ++k) wcb[ch][k] = sgn8(wconv[ch * 16 + k]);
    }

    __syncthreads();

    // -------- Phase B: 4x4 VALID conv + bias + sign -> hb rows in LDS -------
    for (int idx = tid; idx < IMG_PER_BLK * 625; idx += BLK_THREADS) {
        int img = idx / 625;
        int p   = idx - img * 625;
        int r   = p / 25;
        int c   = p - r * 25;
        const signed char* base = xsh + img * XBYTES + r * 28 + c;
        int s0 = 0, s1 = 0, s2 = 0;
#pragma unroll
        for (int i = 0; i < 4; ++i) {
#pragma unroll
            for (int j = 0; j < 4; ++j) {
                int xv = base[i * 28 + j];
                s0 += xv * wcb[0][i * 4 + j];
                s1 += xv * wcb[1][i * 4 + j];
                s2 += xv * wcb[2][i * 4 + j];
            }
        }
        signed char* hrow = hsh + img * HSTRIDE;
        hrow[p]        = sgn8((float)s0 + bc[0]);
        hrow[625 + p]  = sgn8((float)s1 + bc[1]);
        hrow[1250 + p] = sgn8((float)s2 + bc[2]);
    }
    // Zero the K padding [1875, 1920) so WMMA accumulates exactly.
    for (int i = tid; i < IMG_PER_BLK * (K_PAD - K_REAL); i += BLK_THREADS) {
        int img = i / (K_PAD - K_REAL);
        int o   = i - img * (K_PAD - K_REAL);
        hsh[img * HSTRIDE + K_REAL + o] = 0;
    }

    __syncthreads();

    // -------- Phase C: FC GEMM via V_WMMA_I32_16X16X64_IU8 ------------------
    const int wv    = tid >> 5;          // wave id: 0..3, owns 16 images
    const int lane  = tid & 31;
    const int mrow  = lane & 15;         // A-matrix row for this lane
    const int half  = lane >> 4;         // K-half select (0 -> K+0, 1 -> K+8)

    const int* hrowi = (const int*)(hsh + (wv * 16 + mrow) * HSTRIDE);

    v8i acc = {};
    for (int kk = 0; kk < KSTEPS; ++kk) {
        v8i a, b;
        const int kb = kk * 16 + half * 2;   // dword index: (kk*64 + half*8)/4
#pragma unroll
        for (int g = 0; g < 4; ++g) {
            a[2 * g]     = hrowi[kb + g * 4];       // bytes K = kk*64+g*16+half*8
            a[2 * g + 1] = hrowi[kb + g * 4 + 1];   // +4 bytes
        }
        const int* bp = wsB + kk * 256 + lane * 8;
#pragma unroll
        for (int j = 0; j < 8; ++j) b[j] = bp[j];
        // signed(+/-1 int8) x signed(+/-1 int8) -> i32 16x16 accumulators
        acc = __builtin_amdgcn_wmma_i32_16x16x64_iu8(true, a, true, b, acc,
                                                     false, false);
    }

    // Epilogue: C/D layout -> lane col N = lane%16, row M = vgpr + 8*(lane/16)
    const int Ncol = lane & 15;
    if (Ncol < 10) {
        const float bb = bfc[Ncol];
#pragma unroll
        for (int v = 0; v < 8; ++v) {
            int  M   = v + 8 * half;
            long row = blockImg + wv * 16 + M;
            if (row < nImg) {
                float o = (float)acc[v] + bb;
                out[row * 10 + Ncol] = sgnf(o);
            }
        }
    }
}

extern "C" void kernel_launch(void* const* d_in, const int* in_sizes, int n_in,
                              void* d_out, int out_size, void* d_ws, size_t ws_size,
                              hipStream_t stream) {
    const float* x     = (const float*)d_in[0];   // [B,1,28,28]
    const float* wconv = (const float*)d_in[1];   // [3,1,4,4]
    const float* bconv = (const float*)d_in[2];   // [3]
    const float* wfc   = (const float*)d_in[3];   // [10,1875]
    const float* bfc   = (const float*)d_in[4];   // [10]
    float*       out   = (float*)d_out;           // [B,10]
    int*         wsB   = (int*)d_ws;              // 30*32*8 ints = 30 KB

    const int nImg = in_sizes[0] / XBYTES;        // 65536

    // Pack binarized FC weights into WMMA B-fragment layout.
    bnn_pack_wfc<<<30, 256, 0, stream>>>(wfc, wsB);

    // Fused binarize -> conv -> WMMA FC -> sign.
    const int    grid  = (nImg + IMG_PER_BLK - 1) / IMG_PER_BLK;
    const size_t shmem = (size_t)IMG_PER_BLK * HSTRIDE + (size_t)IMG_PER_BLK * XBYTES;
    bnn_fused<<<grid, BLK_THREADS, shmem, stream>>>(x, wconv, bconv, bfc, wsB,
                                                    out, nImg);
}
